// GNNLayer_71640054497691
// MI455X (gfx1250) — compile-verified, matrix operand order checked by reading, not run on hardware
//
#include <hip/hip_runtime.h>
#include <hip/hip_bf16.h>

#define BB 2
#define SS 4
#define NN 256
#define EE 65536
#define HH 64
#define CHUNK 1024
#define TILE 128

typedef _Float16 half16 __attribute__((ext_vector_type(16)));
typedef float    float8 __attribute__((ext_vector_type(8)));
typedef unsigned int uintv4 __attribute__((ext_vector_type(4)));

union Frag { half16 h; uintv4 q[2]; };

#define LOG2E 1.44269504088896340736f
#define LN2   0.69314718055994530942f

// fast softplus: max(x,0) + ln2 * log2(1 + exp2(-|x|*log2e))
// uses native v_exp_f32 / v_log_f32 transcendentals (~7 VALU ops)
__device__ __forceinline__ float softplusf(float x) {
  const float e = __builtin_amdgcn_exp2f(fabsf(x) * -LOG2E);
  return fmaxf(x, 0.f) + LN2 * __builtin_amdgcn_logf(1.f + e);
}
// fast sigmoid: v_rcp_f32(1 + exp2(-x*log2e))
__device__ __forceinline__ float sigmoidf(float x) {
  return __builtin_amdgcn_rcpf(1.f + __builtin_amdgcn_exp2f(x * -LOG2E));
}
// exact float atomic-max via signed/unsigned integer ordering trick
__device__ __forceinline__ void atomicMaxF(float* addr, float val) {
  if (val >= 0.f) atomicMax((int*)addr, __float_as_int(val));
  else            atomicMin((unsigned int*)addr, __float_as_uint(val));
}

struct EdgeParams {
  const float* t; const float* ef; const int* erow; const int* emask;
  const float* wx0; const float* bx0; const float* wt0; const float* wbt0;
  const float* wx1; const float* bx1; const float* wt1; const float* wbt1;
  const float* wx2; const float* bx2; const float* wt2; const float* wbt2;
  const float* wx3; const float* bx3; const float* wt3; const float* wbt3;
  const float* wx4; const float* bx4; const float* wt4; const float* wbt4;
  const float* wx5; const float* bx5; const float* wt5; const float* wbt5;
  float* gl; float* p4; float* smax; float* mfull;
};

// ---------------------------------------------------------------- init ------
__global__ __launch_bounds__(256) void init_kernel(float* smax, float* ssum,
                                                   float* xnum, float* mfull) {
  const int i = blockIdx.x * 256 + threadIdx.x;
  if (i < BB * SS * NN) { smax[i] = -__builtin_inff(); ssum[i] = 0.f; }
  if (i < BB * SS * NN * 3) xnum[i] = 0.f;
  if (i < BB * SS * NN * HH) mfull[i] = 0.f;
}

// ---------------------------------------------------------------- edges -----
__global__ __launch_bounds__(256, 1) void edge_kernel(EdgeParams P) {
  __shared__ _Float16 sX[TILE * 72];        // activations, [edge][feat], pad 72
  __shared__ _Float16 sW[3][64 * 72];       // W1,W2,W3 transposed [n][k], pad 72
  __shared__ float sB[3][64], sSig[3][64], sHb[3][64];
  __shared__ float sWx0[64], sWx4[64], sWx5[64];
  __shared__ float sEm[TILE], sG0[TILE], sP4a[TILE], sS5p[TILE], sS5[TILE];
  __shared__ int   sRow[TILE];
  __shared__ float sc[9];

  const int tid   = threadIdx.x;
  const int bs    = blockIdx.x / (EE / CHUNK);
  const int chunk = blockIdx.x % (EE / CHUNK);
  const float tval = P.t[bs];

  { // -------- prologue: stage weights / per-(b,s) gate vectors into LDS -----
    const float* wsrc[3]  = {P.wx1, P.wx2, P.wx3};
    const float* bsrc[3]  = {P.bx1, P.bx2, P.bx3};
    const float* tsrc[3]  = {P.wt1, P.wt2, P.wt3};
    const float* btsrc[3] = {P.wbt1, P.wbt2, P.wbt3};
#pragma unroll
    for (int l = 0; l < 3; ++l) {
      for (int i = tid; i < 4096; i += 256) {
        const int n = i >> 6, k = i & 63;
        sW[l][n * 72 + k] = (_Float16)wsrc[l][k * 64 + n];   // transpose
      }
      for (int i = tid; i < 64; i += 256) {
        sB[l][i]   = bsrc[l][i];
        sSig[l][i] = sigmoidf(tval * tsrc[l][i]);
        sHb[l][i]  = tval * btsrc[l][i];
      }
    }
    for (int i = tid; i < 64; i += 256) {
      sWx0[i] = P.wx0[i]; sWx4[i] = P.wx4[i]; sWx5[i] = P.wx5[i];
    }
    if (tid == 0) {
      sc[0] = P.bx0[0]; sc[1] = sigmoidf(tval * P.wt0[0]); sc[2] = tval * P.wbt0[0];
      sc[3] = P.bx4[0]; sc[4] = sigmoidf(tval * P.wt4[0]); sc[5] = tval * P.wbt4[0];
      sc[6] = P.bx5[0]; sc[7] = sigmoidf(tval * P.wt5[0]); sc[8] = tval * P.wbt5[0];
    }
  }
  __syncthreads();

  const int lane = tid & 31;
  const int wv   = tid >> 5;          // wave id == edge M-block
  const int g    = (lane >> 4) & 1;
  const int r    = lane & 15;

  for (int tl = 0; tl < CHUNK / TILE; ++tl) {
    const int    ebase = chunk * CHUNK + tl * TILE;
    const size_t gbase = (size_t)bs * EE + ebase;

    if (tid < TILE) { sG0[tid] = 0.f; sP4a[tid] = 0.f; sS5p[tid] = 0.f; }
    __syncthreads();

    { // -------- load tile: x = ef*em -> f16 LDS, fuse gate dot (W0) ---------
      const int e = tid >> 1, hf = tid & 1;
      const float em = (float)P.emask[gbase + e];
      if (hf == 0) sEm[e] = em;
      const float4* src = (const float4*)(P.ef + (gbase + e) * HH + hf * 32);
      float partial = 0.f;
#pragma unroll
      for (int j = 0; j < 8; ++j) {
        const float4 v = src[j];
        const int f0 = hf * 32 + j * 4;
        const float x0 = v.x * em, x1 = v.y * em, x2 = v.z * em, x3 = v.w * em;
        partial += x0 * sWx0[f0] + x1 * sWx0[f0 + 1] + x2 * sWx0[f0 + 2] + x3 * sWx0[f0 + 3];
        sX[e * 72 + f0]     = (_Float16)x0;  sX[e * 72 + f0 + 1] = (_Float16)x1;
        sX[e * 72 + f0 + 2] = (_Float16)x2;  sX[e * 72 + f0 + 3] = (_Float16)x3;
      }
      atomicAdd(&sG0[e], partial);
    }
    __syncthreads();

    // -------- three WMMA layers: W1, W2 (out*em), W3 (fused W4 dot) ---------
#pragma unroll
    for (int layer = 0; layer < 3; ++layer) {
      Frag a0, a1;
      const int abase = (wv * 16 + r) * 72 + g * 8;    // documented 16b A layout
      a0.q[0] = *(const uintv4*)&sX[abase];
      a0.q[1] = *(const uintv4*)&sX[abase + 16];
      a1.q[0] = *(const uintv4*)&sX[abase + 32];
      a1.q[1] = *(const uintv4*)&sX[abase + 48];
      __syncthreads();                                  // everyone holds A
#pragma unroll
      for (int nb = 0; nb < 4; ++nb) {
        Frag b0, b1;
        const int wbase = (nb * 16 + r) * 72 + g * 8;
        const _Float16* Wl = sW[layer];
        b0.q[0] = *(const uintv4*)&Wl[wbase];
        b0.q[1] = *(const uintv4*)&Wl[wbase + 16];
        b1.q[0] = *(const uintv4*)&Wl[wbase + 32];
        b1.q[1] = *(const uintv4*)&Wl[wbase + 48];
        float8 acc = {};
        acc = __builtin_amdgcn_wmma_f32_16x16x32_f16(false, a0.h, false, b0.h,
                                                     (short)0, acc, false, false);
        acc = __builtin_amdgcn_wmma_f32_16x16x32_f16(false, a1.h, false, b1.h,
                                                     (short)0, acc, false, false);
        const int nf = nb * 16 + r;
        const float bb = sB[layer][nf], sg = sSig[layer][nf], hb = sHb[layer][nf];
#pragma unroll
        for (int r8 = 0; r8 < 8; ++r8) {                // C layout: M = g*8+r8
          const int edge = wv * 16 + g * 8 + r8;
          float y = softplusf((acc[r8] + bb) * sg + hb);
          if (layer == 0) {
            sX[edge * 72 + nf] = (_Float16)y;           // h1
          } else if (layer == 1) {
            y *= sEm[edge];                             // out = h2 * em
            atomicAdd(&sS5p[edge], y * sWx5[nf]);       // fused W5 dot
            sX[edge * 72 + nf] = (_Float16)y;
          } else {
            atomicAdd(&sP4a[edge], y * sWx4[nf]);       // fused W4 dot (p3)
          }
        }
      }
      __syncthreads();
    }

    // -------- epilogue A: per-edge scalars, smax atomic-max -----------------
    if (tid < TILE) {
      const int e = tid;
      const size_t gi = gbase + e;
      const float em = sEm[e];
      const float gl = (em > 0.f) ? ((sG0[e] + sc[0]) * sc[1] + sc[2]) : -1e20f;
      P.gl[gi] = gl;
      P.p4[gi] = (sP4a[e] + sc[3]) * sc[4] + sc[5];
      sS5[e]   = sigmoidf((sS5p[e] + sc[6]) * sc[7] + sc[8]);
      const int row = P.erow[ebase + e];
      sRow[e] = row;
      atomicMaxF(&P.smax[bs * NN + row], gl);
    }
    __syncthreads();

    // -------- epilogue B: scatter m = sigmoid(cs5)*out into node features ---
#pragma unroll 4
    for (int i = 0; i < 32; ++i) {
      const int idx = tid + i * 256;
      const int e = idx >> 6, f = idx & 63;
      if (sEm[e] > 0.f) {
        const float val = sS5[e] * (float)sX[e * 72 + f];
        atomicAdd(&P.mfull[((size_t)bs * NN + sRow[e]) * HH + f], val);
      }
    }
    __syncthreads();
  }
}

// ---------------------------------------------------------------- softmax x -
__global__ __launch_bounds__(256) void scatter_x_kernel(
    const float* gl, const float* p4, const float* smax,
    const float* diff, const int* erow, float* ssum, float* xnum) {
  const size_t i = (size_t)blockIdx.x * 256 + threadIdx.x;   // < B*S*E
  const int bs = (int)(i >> 16);                             // E == 2^16
  const int e  = (int)(i & (EE - 1));
  const int node = bs * NN + erow[e];
  const float w = __builtin_amdgcn_exp2f((gl[i] - smax[node]) * LOG2E);
  atomicAdd(&ssum[node], w);
  const float pw = p4[i] * w;
  const float dx = diff[i * 3], dy = diff[i * 3 + 1], dz = diff[i * 3 + 2];
  const float inv = pw * __builtin_amdgcn_rcpf(
      sqrtf(dx * dx + dy * dy + dz * dz + 3e-12f) + 1.f);
  atomicAdd(&xnum[node * 3 + 0], inv * dx);
  atomicAdd(&xnum[node * 3 + 1], inv * dy);
  atomicAdd(&xnum[node * 3 + 2], inv * dz);
}

// ---------------------------------------------------------------- nodes -----
__global__ __launch_bounds__(256) void node_kernel(
    const float* t, const float* atomf, const float* mfull,
    const float* ssum, const float* xnum, const int* amask,
    const float* wx6, const float* bx6, const float* wt6, const float* wbt6,
    const float* wx7, const float* bx7, const float* wt7, const float* wbt7,
    float* out) {
  __shared__ float h6s[8][64];
  const int tid = threadIdx.x, lane = tid & 31, wv = tid >> 5;
  const int node = blockIdx.x * 8 + wv;                      // < B*S*N
  const int bs = node / NN, n = node % NN, b = bs / SS;
  const float tval = t[bs];
  const float* af = atomf + (size_t)node * HH;
  const float* mf = mfull + (size_t)node * HH;
#pragma unroll
  for (int h = 0; h < 2; ++h) {
    const int f = lane + h * 32;
    float s = 0.f;
    for (int k = 0; k < 64; ++k) s += af[k] * wx6[k * 64 + f];
    for (int k = 0; k < 64; ++k) s += mf[k] * wx6[(64 + k) * 64 + f];
    const float y = (s + bx6[f]) * sigmoidf(tval * wt6[f]) + tval * wbt6[f];
    h6s[wv][f] = softplusf(y);
  }
  __syncthreads();
#pragma unroll
  for (int h = 0; h < 2; ++h) {
    const int f = lane + h * 32;
    float s = 0.f;
    for (int k = 0; k < 64; ++k) s += h6s[wv][k] * wx7[k * 64 + f];
    const float y = (s + bx7[f]) * sigmoidf(tval * wt7[f]) + tval * wbt7[f];
    out[BB * SS * NN * 3 + (size_t)node * HH + f] = y + af[f];
  }
  if (lane < 3) {
    const float ssv = ssum[node];
    const float x = (ssv > 0.f) ? (xnum[node * 3 + lane] / ssv) : 0.f;
    out[(size_t)node * 3 + lane] = x * (float)amask[b * NN + n];
  }
}

// ---------------------------------------------------------------- launcher --
extern "C" void kernel_launch(void* const* d_in, const int* in_sizes, int n_in,
                              void* d_out, int out_size, void* d_ws, size_t ws_size,
                              hipStream_t stream) {
  (void)in_sizes; (void)n_in; (void)out_size; (void)ws_size;
  const float* t     = (const float*)d_in[0];
  const float* atomf = (const float*)d_in[1];
  const float* diff  = (const float*)d_in[2];
  const float* ef    = (const float*)d_in[3];
  const int*   erow  = (const int*)d_in[4];
  const int*   emask = (const int*)d_in[5];
  const int*   amask = (const int*)d_in[6];
  auto Wp = [&](int i, int j) { return (const float*)d_in[7 + 4 * i + j]; };

  float* ws = (float*)d_ws;
  const size_t BSE = (size_t)BB * SS * EE;
  float* gl    = ws;
  float* p4    = gl + BSE;
  float* smax  = p4 + BSE;
  float* ssum  = smax + BB * SS * NN;
  float* xnum  = ssum + BB * SS * NN;
  float* mfull = xnum + BB * SS * NN * 3;

  hipLaunchKernelGGL(init_kernel, dim3((BB * SS * NN * HH) / 256), dim3(256), 0,
                     stream, smax, ssum, xnum, mfull);

  EdgeParams P;
  P.t = t; P.ef = ef; P.erow = erow; P.emask = emask;
  P.wx0 = Wp(0, 0); P.bx0 = Wp(0, 1); P.wt0 = Wp(0, 2); P.wbt0 = Wp(0, 3);
  P.wx1 = Wp(1, 0); P.bx1 = Wp(1, 1); P.wt1 = Wp(1, 2); P.wbt1 = Wp(1, 3);
  P.wx2 = Wp(2, 0); P.bx2 = Wp(2, 1); P.wt2 = Wp(2, 2); P.wbt2 = Wp(2, 3);
  P.wx3 = Wp(3, 0); P.bx3 = Wp(3, 1); P.wt3 = Wp(3, 2); P.wbt3 = Wp(3, 3);
  P.wx4 = Wp(4, 0); P.bx4 = Wp(4, 1); P.wt4 = Wp(4, 2); P.wbt4 = Wp(4, 3);
  P.wx5 = Wp(5, 0); P.bx5 = Wp(5, 1); P.wt5 = Wp(5, 2); P.wbt5 = Wp(5, 3);
  P.gl = gl; P.p4 = p4; P.smax = smax; P.mfull = mfull;

  hipLaunchKernelGGL(edge_kernel, dim3(BB * SS * (EE / CHUNK)), dim3(256), 0,
                     stream, P);
  hipLaunchKernelGGL(scatter_x_kernel, dim3((unsigned)(BSE / 256)), dim3(256), 0,
                     stream, gl, p4, smax, diff, erow, ssum, xnum);
  hipLaunchKernelGGL(node_kernel, dim3((BB * SS * NN) / 8), dim3(256), 0, stream,
                     t, atomf, mfull, ssum, xnum, amask,
                     Wp(6, 0), Wp(6, 1), Wp(6, 2), Wp(6, 3),
                     Wp(7, 0), Wp(7, 1), Wp(7, 2), Wp(7, 3), (float*)d_out);
}
